// KMeans_2482491097322
// MI455X (gfx1250) — matrix operand order repaired and checked by reading, not compiled, not run on hardware
//
#include <hip/hip_runtime.h>

typedef __attribute__((ext_vector_type(16))) _Float16 v16h;
typedef __attribute__((ext_vector_type(8)))  float    v8f;

#define KC   256   // clusters
#define DIM  64    // dims
#define NITER 10

// ---------------------------------------------------------------- prep points
__global__ __launch_bounds__(256)
void prep_points(const float* __restrict__ x, _Float16* __restrict__ xh,
                 float* __restrict__ xsq, int N) {
  int i = blockIdx.x * blockDim.x + threadIdx.x;
  if (i >= N) return;
  const float* row = x + (size_t)i * DIM;
  float s = 0.f;
#pragma unroll
  for (int d = 0; d < DIM; ++d) {
    float v = row[d];
    s += v * v;
    xh[(size_t)i * DIM + d] = (_Float16)v;
  }
  xsq[i] = s;
}

// ---------------------------------------------------------------- init c = x[:K]
__global__ __launch_bounds__(256)
void init_centroids(const float* __restrict__ x, float* __restrict__ c) {
  int i = blockIdx.x * blockDim.x + threadIdx.x;
  if (i < KC * DIM) c[i] = x[i];
}

// ---------------------------------------------------------------- per-iter prep
__global__ __launch_bounds__(256)
void prep_centroids(const float* __restrict__ c, _Float16* __restrict__ ch,
                    float* __restrict__ csq) {
  int k = blockIdx.x * blockDim.x + threadIdx.x;
  if (k >= KC) return;
  const float* row = c + k * DIM;
  float s = 0.f;
#pragma unroll
  for (int d = 0; d < DIM; ++d) {
    float v = row[d];
    s += v * v;
    ch[k * DIM + d] = (_Float16)v;
  }
  csq[k] = s;
}

__global__ __launch_bounds__(256)
void zero_accum(float* __restrict__ gsums, float* __restrict__ gcnt) {
  int i = blockIdx.x * blockDim.x + threadIdx.x;
  if (i < KC * DIM) gsums[i] = 0.f;
  if (i < KC) gcnt[i] = 0.f;
}

// ---------------------------------------------------------------- assignment + LDS segment sums
// 128 threads = 4 waves; each wave owns a 32-point tile (two 16-row A pairs)
// per pass, so every B-fragment LDS read feeds 4 WMMAs.
__global__ __launch_bounds__(128)
void assign_kernel(const _Float16* __restrict__ xh, const float* __restrict__ x,
                   const float* __restrict__ xsq,
                   const _Float16* __restrict__ ch, const float* __restrict__ csq,
                   int* __restrict__ labels, float* __restrict__ gsums,
                   float* __restrict__ gcnt, int pointsPerBlock) {
  __shared__ _Float16 sC[KC * DIM];    // 32 KB centroid tile (f16)
  __shared__ float    sCsq[KC];        //  1 KB
  __shared__ float    sSum[KC * DIM];  // 64 KB fp32 accumulator
  __shared__ float    sCnt[KC];        //  1 KB

  const int tid = threadIdx.x;

  // cooperative centroid load (16B vectors) + accumulator zeroing
  {
    const uint4* src = (const uint4*)ch;
    uint4* dst = (uint4*)sC;
    for (int i = tid; i < KC * DIM / 8; i += 128) dst[i] = src[i];
    for (int i = tid; i < KC; i += 128) { sCsq[i] = csq[i]; sCnt[i] = 0.f; }
    for (int i = tid; i < KC * DIM; i += 128) sSum[i] = 0.f;
  }
  __syncthreads();

  const int lane   = tid & 31;
  const int wave   = tid >> 5;
  const int lane16 = lane & 15;
  const int hi     = lane >> 4;   // which half of the wave

  const int blockBase = blockIdx.x * pointsPerBlock;

  for (int base = 0; base < pointsPerBlock; base += 128) {
    const int tile = blockBase + base + wave * 32;   // 32 points for this wave

    // ---- A fragments, ISA 16-bit A 16x32 lane layout, 32-bit pair loads.
    // a0/a1: rows tile..tile+15, dims 0-31 / 32-63.  a2/a3: rows tile+16..tile+31.
    union { v16h h; unsigned u[8]; } a0, a1, a2, a3;
    const _Float16* xr0 = xh + (size_t)(tile + lane16) * DIM;
    const _Float16* xr1 = xh + (size_t)(tile + 16 + lane16) * DIM;
#pragma unroll
    for (int v = 0; v < 8; ++v) {
      int k = (v < 4) ? (2 * v + (hi ? 8 : 0)) : (2 * (v - 4) + (hi ? 24 : 16));
      a0.u[v] = *(const unsigned*)(xr0 + k);
      a1.u[v] = *(const unsigned*)(xr0 + 32 + k);
      a2.u[v] = *(const unsigned*)(xr1 + k);
      a3.u[v] = *(const unsigned*)(xr1 + 32 + k);
    }

    // per-lane ||x||^2 for the 8+8 rows this lane's C/D registers cover
    float xs0[8], xs1[8];
#pragma unroll
    for (int r = 0; r < 8; ++r) {
      xs0[r] = xsq[tile + hi * 8 + r];
      xs1[r] = xsq[tile + 16 + hi * 8 + r];
    }

    float minv0[8], minv1[8];
    int   mini0[8], mini1[8];
#pragma unroll
    for (int r = 0; r < 8; ++r) {
      minv0[r] = 3.4e38f; mini0[r] = 0;
      minv1[r] = 3.4e38f; mini1[r] = 0;
    }

    // ---- sweep 16 cluster tiles of 16; each B load feeds 4 WMMAs
#pragma unroll 2
    for (int t = 0; t < 16; ++t) {
      const int n0 = t * 16;
      union { v16h h; unsigned u[8]; } b0, b1;
      const _Float16* cr = sC + (n0 + lane16) * DIM;
#pragma unroll
      for (int v = 0; v < 8; ++v) {
        int k = (v < 4) ? (2 * v + (hi ? 8 : 0)) : (2 * (v - 4) + (hi ? 24 : 16));
        b0.u[v] = *(const unsigned*)(cr + k);
        b1.u[v] = *(const unsigned*)(cr + 32 + k);
      }
      v8f acc0 = {}, acc1 = {};
      acc0 = __builtin_amdgcn_wmma_f32_16x16x32_f16(false, a0.h, false, b0.h,
                                                    (short)0, acc0, false, false);
      acc1 = __builtin_amdgcn_wmma_f32_16x16x32_f16(false, a2.h, false, b0.h,
                                                    (short)0, acc1, false, false);
      acc0 = __builtin_amdgcn_wmma_f32_16x16x32_f16(false, a1.h, false, b1.h,
                                                    (short)0, acc0, false, false);
      acc1 = __builtin_amdgcn_wmma_f32_16x16x32_f16(false, a3.h, false, b1.h,
                                                    (short)0, acc1, false, false);
      const int n = n0 + lane16;
      const float cs = sCsq[n];
#pragma unroll
      for (int r = 0; r < 8; ++r) {
        float d0 = xs0[r] - 2.0f * acc0[r] + cs;
        float d1 = xs1[r] - 2.0f * acc1[r] + cs;
        bool t0 = d0 < minv0[r];
        bool t1 = d1 < minv1[r];
        minv0[r] = t0 ? d0 : minv0[r];
        mini0[r] = t0 ? n  : mini0[r];
        minv1[r] = t1 ? d1 : minv1[r];
        mini1[r] = t1 ? n  : mini1[r];
      }
    }

    // ---- branchless argmin across the 16 lanes of each half
#pragma unroll
    for (int m = 1; m <= 8; m <<= 1) {
#pragma unroll
      for (int r = 0; r < 8; ++r) {
        float ov0 = __shfl_xor(minv0[r], m, 32);
        int   oi0 = __shfl_xor(mini0[r], m, 32);
        float ov1 = __shfl_xor(minv1[r], m, 32);
        int   oi1 = __shfl_xor(mini1[r], m, 32);
        bool b0 = (ov0 < minv0[r]) | ((ov0 == minv0[r]) & (oi0 < mini0[r]));
        bool b1 = (ov1 < minv1[r]) | ((ov1 == minv1[r]) & (oi1 < mini1[r]));
        minv0[r] = b0 ? ov0 : minv0[r];
        mini0[r] = b0 ? oi0 : mini0[r];
        minv1[r] = b1 ? ov1 : minv1[r];
        mini1[r] = b1 ? oi1 : mini1[r];
      }
    }
    if (lane16 == 0) {
#pragma unroll
      for (int r = 0; r < 8; ++r) {
        labels[tile + hi * 8 + r]      = mini0[r];
        labels[tile + 16 + hi * 8 + r] = mini1[r];
      }
    }

    // ---- segment-sum the 32 points into the LDS accumulator (fp32 source data)
#pragma unroll
    for (int q = 0; q < 16; ++q) {
      int lab0 = __shfl(mini0[q & 7], (q >> 3) * 16, 32);
      int lab1 = __shfl(mini1[q & 7], (q >> 3) * 16, 32);
      const float* xp0 = x + (size_t)(tile + q) * DIM + lane * 2;
      const float* xp1 = x + (size_t)(tile + 16 + q) * DIM + lane * 2;
      float2 xv0 = *(const float2*)xp0;
      float2 xv1 = *(const float2*)xp1;
      __hip_atomic_fetch_add(&sSum[lab0 * DIM + lane * 2],     xv0.x,
                             __ATOMIC_RELAXED, __HIP_MEMORY_SCOPE_WORKGROUP);
      __hip_atomic_fetch_add(&sSum[lab0 * DIM + lane * 2 + 1], xv0.y,
                             __ATOMIC_RELAXED, __HIP_MEMORY_SCOPE_WORKGROUP);
      __hip_atomic_fetch_add(&sSum[lab1 * DIM + lane * 2],     xv1.x,
                             __ATOMIC_RELAXED, __HIP_MEMORY_SCOPE_WORKGROUP);
      __hip_atomic_fetch_add(&sSum[lab1 * DIM + lane * 2 + 1], xv1.y,
                             __ATOMIC_RELAXED, __HIP_MEMORY_SCOPE_WORKGROUP);
      if (lane == q)
        __hip_atomic_fetch_add(&sCnt[lab0], 1.0f,
                               __ATOMIC_RELAXED, __HIP_MEMORY_SCOPE_WORKGROUP);
      if (lane == q + 16)
        __hip_atomic_fetch_add(&sCnt[lab1], 1.0f,
                               __ATOMIC_RELAXED, __HIP_MEMORY_SCOPE_WORKGROUP);
    }
  }

  __syncthreads();
  // single global flush per block
  for (int i = tid; i < KC * DIM; i += 128)
    __hip_atomic_fetch_add(&gsums[i], sSum[i],
                           __ATOMIC_RELAXED, __HIP_MEMORY_SCOPE_AGENT);
  for (int i = tid; i < KC; i += 128)
    __hip_atomic_fetch_add(&gcnt[i], sCnt[i],
                           __ATOMIC_RELAXED, __HIP_MEMORY_SCOPE_AGENT);
}

// ---------------------------------------------------------------- update
__global__ __launch_bounds__(256)
void update_centroids(const float* __restrict__ gsums, const float* __restrict__ gcnt,
                      float* __restrict__ c, float* __restrict__ outc) {
  int i = blockIdx.x * blockDim.x + threadIdx.x;
  if (i >= KC * DIM) return;
  float v = gsums[i] / gcnt[i / DIM];
  c[i] = v;
  outc[i] = v;   // final iteration leaves the answer in d_out
}

// ---------------------------------------------------------------- launch
extern "C" void kernel_launch(void* const* d_in, const int* in_sizes, int n_in,
                              void* d_out, int out_size, void* d_ws, size_t ws_size,
                              hipStream_t stream) {
  const float* x = (const float*)d_in[0];
  const int N = in_sizes[0] / DIM;   // 262144

  float* outc = (float*)d_out;                 // KC*DIM fp32 centroids
  int*   outl = (int*)(outc + KC * DIM);       // N int32 labels

  // workspace layout
  char* ws = (char*)d_ws;
  _Float16* xh   = (_Float16*)ws;                        // N*DIM halves (32 MB)
  float*    xsq  = (float*)(ws + (size_t)N * DIM * 2);   // N
  float*    c    = xsq + N;                              // KC*DIM
  float*    csq  = c + KC * DIM;                         // KC
  _Float16* ch   = (_Float16*)(csq + KC);                // KC*DIM halves
  float*    gsums = (float*)(ch + KC * DIM);             // KC*DIM
  float*    gcnt  = gsums + KC * DIM;                    // KC

  prep_points<<<(N + 255) / 256, 256, 0, stream>>>(x, xh, xsq, N);
  init_centroids<<<(KC * DIM + 255) / 256, 256, 0, stream>>>(x, c);

  const int pointsPerBlock = 512;
  const int nBlocks = N / pointsPerBlock;   // 512 blocks, 4 passes each

  for (int it = 0; it < NITER; ++it) {
    prep_centroids<<<1, 256, 0, stream>>>(c, ch, csq);
    zero_accum<<<(KC * DIM + 255) / 256, 256, 0, stream>>>(gsums, gcnt);
    assign_kernel<<<nBlocks, 128, 0, stream>>>(xh, x, xsq, ch, csq,
                                               outl, gsums, gcnt, pointsPerBlock);
    update_centroids<<<(KC * DIM + 255) / 256, 256, 0, stream>>>(gsums, gcnt, c, outc);
  }
}
